// OnlyAttention_65919158059583
// MI455X (gfx1250) — matrix operand order repaired and checked by reading, not compiled, not run on hardware
//
#include <hip/hip_runtime.h>
#include <hip/hip_bf16.h>

#define B_  8
#define T_  2048
#define D_  1024
#define DO_ 1024

typedef __attribute__((ext_vector_type(16))) _Float16 v16h;
typedef __attribute__((ext_vector_type(8)))  _Float16 h8;
typedef __attribute__((ext_vector_type(8)))  float    v8f;
typedef __attribute__((ext_vector_type(4)))  int      i32x4;

typedef __attribute__((address_space(1))) i32x4 gv4i;   // global int4
typedef __attribute__((address_space(3))) i32x4 lv4i;   // LDS int4

#if defined(__has_builtin)
#  if __has_builtin(__builtin_amdgcn_global_load_async_to_lds_b128)
#    define HAVE_ASYNC_LDS 1
#  endif
#endif
#ifndef HAVE_ASYNC_LDS
#  define HAVE_ASYNC_LDS 0
#endif

__device__ __forceinline__ void wait_asynccnt0() {
#if defined(__has_builtin)
#  if __has_builtin(__builtin_amdgcn_s_wait_asynccnt)
    __builtin_amdgcn_s_wait_asynccnt(0);
    return;
#  endif
#endif
    asm volatile("s_wait_asynccnt 0" ::: "memory");
}

// ---------------------------------------------------------------------------
// A-fragment (16x32 f16, M x K): lane holds row M = lane&15;
// half h -> K = k0 + (h>>3)*16 + (lane>>4)*8 + (h&7)
// => two contiguous 16B chunks at k0+(lane>>4)*8 and +16 further.
// ---------------------------------------------------------------------------
__device__ __forceinline__ v16h load_a_frag(const _Float16* base, int stride,
                                            int row_base, int k0, int lane) {
    const int m    = lane & 15;
    const int koff = k0 + ((lane >> 4) << 3);
    const _Float16* p = base + (long)(row_base + m) * stride + koff;
    h8 lo = *(const h8*)(p);
    h8 hi = *(const h8*)(p + 16);
    v16h a;
#pragma unroll
    for (int i = 0; i < 8; ++i) { a[i] = lo[i]; a[i + 8] = hi[i]; }
    return a;
}

// ---------------------------------------------------------------------------
// B-fragment (32x16 f16, K x N): lane holds column N = lane&15;
// half h -> K = k0 + (h>>3)*16 + (lane>>4)*8 + (h&7)
// Source laid out [n][k] row-major.
// ---------------------------------------------------------------------------
__device__ __forceinline__ v16h load_b_frag(const _Float16* base, int stride,
                                            int col_base, int k0, int lane) {
    const int n    = lane & 15;
    const int koff = k0 + ((lane >> 4) << 3);
    const _Float16* p = base + (long)(col_base + n) * stride + koff;
    h8 lo = *(const h8*)(p);
    h8 hi = *(const h8*)(p + 16);
    v16h b;
#pragma unroll
    for (int i = 0; i < 8; ++i) { b[i] = lo[i]; b[i + 8] = hi[i]; }
    return b;
}

// ---------------------------------------------------------------------------
// fp32 -> f16 elementwise convert, 8 elements / thread
// ---------------------------------------------------------------------------
__global__ void cvt_f32_to_f16(const float* __restrict__ in,
                               _Float16* __restrict__ out, int n) {
    int idx = (blockIdx.x * blockDim.x + threadIdx.x) * 8;
    if (idx + 8 <= n) {
        h8 v;
#pragma unroll
        for (int i = 0; i < 8; ++i) v[i] = (_Float16)in[idx + i];
        *(h8*)(out + idx) = v;
    }
}

// fp32 [K][N] -> f16 [N][K] transpose-convert (coalesced writes)
__global__ void cvt_transpose_f32_to_f16(const float* __restrict__ in,
                                         _Float16* __restrict__ out,
                                         int K, int N) {
    int idx = blockIdx.x * blockDim.x + threadIdx.x;
    if (idx < K * N) {
        int n = idx / K;
        int k = idx - n * K;
        out[idx] = (_Float16)in[(long)k * N + n];
    }
}

// ---------------------------------------------------------------------------
// Projection GEMM: Y = (Xh @ W + bias) * scale   (W given transposed [N][K])
// One 16x128 output STRIP per wave (8 accumulators) -> each A-fragment feeds
// 8 WMMAs. 4 waves / block.
// transposeStore: write Y transposed per batch as Yt[b][n][t] (for V).
// ---------------------------------------------------------------------------
__global__ void __launch_bounds__(128)
proj_gemm_f16(const _Float16* __restrict__ Xh, const _Float16* __restrict__ Wt,
              const float* __restrict__ bias, _Float16* __restrict__ Y,
              int K, int N, float scale, int transposeStore) {
    const int lane = threadIdx.x & 31;
    const int wid  = threadIdx.x >> 5;
    const int strip = blockIdx.x * 4 + wid;
    const int stripsPerRow = N >> 7;                 // strips of 128 cols
    const int rowTile = strip / stripsPerRow;
    const int n0 = (strip - rowTile * stripsPerRow) << 7;
    const int r0 = rowTile << 4;

    v8f acc[8];
#pragma unroll
    for (int nt = 0; nt < 8; ++nt) acc[nt] = (v8f){};

    for (int k0 = 0; k0 < K; k0 += 32) {
        v16h a = load_a_frag(Xh, K, r0, k0, lane);
#pragma unroll
        for (int nt = 0; nt < 8; ++nt) {
            v16h b = load_b_frag(Wt, K, n0 + (nt << 4), k0, lane);
            acc[nt] = __builtin_amdgcn_wmma_f32_16x16x32_f16(false, a, false, b,
                                                             (short)0, acc[nt], false, false);
        }
    }

    if (transposeStore) {
        const int b    = r0 >> 11;                   // r0 / T
        const int tloc = (r0 & (T_ - 1)) + ((lane >> 4) << 3);
#pragma unroll
        for (int nt = 0; nt < 8; ++nt) {
            const int o  = n0 + (nt << 4) + (lane & 15);
            const float bv = bias[o];
            h8 v;
#pragma unroll
            for (int r = 0; r < 8; ++r) v[r] = (_Float16)((acc[nt][r] + bv) * scale);
            *(h8*)(Y + (long)b * N * T_ + (long)o * T_ + tloc) = v;
        }
    } else {
        const int mb = r0 + ((lane >> 4) << 3);
#pragma unroll
        for (int nt = 0; nt < 8; ++nt) {
            const int n  = n0 + (nt << 4) + (lane & 15);
            const float bv = bias[n];
#pragma unroll
            for (int r = 0; r < 8; ++r)
                Y[(long)(mb + r) * N + n] = (_Float16)((acc[nt][r] + bv) * scale);
        }
    }
}

// ---------------------------------------------------------------------------
// Attention: one workgroup per (batch, 16-row tile). 8 waves, 256 threads.
// Dynamic LDS: P scores (16 x 2048 f16, 64KB) + Qs tile (16 x 1024 f16, 32KB).
// Phase 0: async-copy Q tile to LDS (ASYNCcnt path when available)
// Phase 1: S = Qs @ K^T, 4 concurrent s-tiles per wave sharing the A-frag
// Phase 2: per-row softmax (in-wave shuffle reductions), P back to LDS
// Phase 3: O = P @ V (8 accumulators per wave, A-frag reuse)
// ---------------------------------------------------------------------------
__global__ void __launch_bounds__(256)
attn_kernel(const _Float16* __restrict__ Qh, const _Float16* __restrict__ Kh,
            const _Float16* __restrict__ Vt, float* __restrict__ out) {
    extern __shared__ _Float16 smem[];
    _Float16* P  = smem;                 // 16 * 2048 halves
    _Float16* Qs = smem + 16 * T_;       // 16 * 1024 halves

    const int lane = threadIdx.x & 31;
    const int wid  = threadIdx.x >> 5;
    const int b    = blockIdx.y;
    const int r0   = blockIdx.x << 4;

    const _Float16* Qb = Qh + (long)b * T_ * D_;
    const _Float16* Kb = Kh + (long)b * T_ * D_;
    const _Float16* Vb = Vt + (long)b * DO_ * T_;

    // ---- Phase 0: stage the 32KB Q tile (16 rows, contiguous) into LDS ----
    {
        const _Float16* src = Qb + (long)r0 * D_;
#if HAVE_ASYNC_LDS
#pragma unroll
        for (int i = 0; i < 8; ++i) {
            const int idx = threadIdx.x * 8 + i * 2048;   // 16B per lane per op
            __builtin_amdgcn_global_load_async_to_lds_b128(
                (gv4i*)(src + idx), (lv4i*)(Qs + idx), 0, 0);
        }
        wait_asynccnt0();
#else
#pragma unroll
        for (int i = 0; i < 8; ++i) {
            const int idx = threadIdx.x * 8 + i * 2048;
            *(h8*)(Qs + idx) = *(const h8*)(src + idx);
        }
#endif
        __syncthreads();
    }

    // ---- Phase 1: score tiles; wave w owns s-tiles [16w, 16w+16) ----
    for (int jg = 0; jg < 4; ++jg) {
        const int sbase = ((wid << 4) + (jg << 2)) << 4;
        v8f acc4[4];
#pragma unroll
        for (int u = 0; u < 4; ++u) acc4[u] = (v8f){};

        for (int k0 = 0; k0 < D_; k0 += 32) {
            // A-fragment of Q from LDS
            const int abase = (lane & 15) * D_ + k0 + ((lane >> 4) << 3);
            h8 alo = *(const h8*)&Qs[abase];
            h8 ahi = *(const h8*)&Qs[abase + 16];
            v16h a;
#pragma unroll
            for (int i = 0; i < 8; ++i) { a[i] = alo[i]; a[i + 8] = ahi[i]; }
#pragma unroll
            for (int u = 0; u < 4; ++u) {
                v16h kb = load_b_frag(Kb, D_, sbase + (u << 4), k0, lane);
                acc4[u] = __builtin_amdgcn_wmma_f32_16x16x32_f16(false, a, false, kb,
                                                                 (short)0, acc4[u], false, false);
            }
        }
#pragma unroll
        for (int u = 0; u < 4; ++u) {
            const int n  = sbase + (u << 4) + (lane & 15);
            const int mb = (lane >> 4) << 3;
#pragma unroll
            for (int r = 0; r < 8; ++r)
                P[(mb + r) * T_ + n] = (_Float16)acc4[u][r];
        }
    }
    __syncthreads();

    // ---- Phase 2: softmax; wave w owns rows 2w and 2w+1 ----
    for (int which = 0; which < 2; ++which) {
        const int row  = (wid << 1) + which;
        const int base = row * T_ + lane * 64;       // 64 contiguous f16 / lane
        float v[64];
#pragma unroll
        for (int c = 0; c < 8; ++c) {
            h8 chunk = *(const h8*)&P[base + c * 8];
#pragma unroll
            for (int i = 0; i < 8; ++i) v[c * 8 + i] = (float)chunk[i];
        }
        float m = -3.402823466e38f;
#pragma unroll
        for (int i = 0; i < 64; ++i) m = fmaxf(m, v[i]);
#pragma unroll
        for (int off = 16; off >= 1; off >>= 1)
            m = fmaxf(m, __shfl_xor(m, off, 32));
        float s = 0.0f;
#pragma unroll
        for (int i = 0; i < 64; ++i) { v[i] = __expf(v[i] - m); s += v[i]; }
#pragma unroll
        for (int off = 16; off >= 1; off >>= 1)
            s += __shfl_xor(s, off, 32);
        const float inv = 1.0f / s;
#pragma unroll
        for (int c = 0; c < 8; ++c) {
            h8 chunk;
#pragma unroll
            for (int i = 0; i < 8; ++i) chunk[i] = (_Float16)(v[c * 8 + i] * inv);
            *(h8*)&P[base + c * 8] = chunk;
        }
    }
    __syncthreads();

    // ---- Phase 3: O = P @ V ; wave w owns output cols [128w, 128w+128) ----
    const int n0 = wid << 7;
    v8f acc[8];
#pragma unroll
    for (int nt = 0; nt < 8; ++nt) acc[nt] = (v8f){};

    for (int s0 = 0; s0 < T_; s0 += 32) {
        const int abase = (lane & 15) * T_ + s0 + ((lane >> 4) << 3);
        h8 alo = *(const h8*)&P[abase];
        h8 ahi = *(const h8*)&P[abase + 16];
        v16h a;
#pragma unroll
        for (int i = 0; i < 8; ++i) { a[i] = alo[i]; a[i + 8] = ahi[i]; }
#pragma unroll
        for (int nt = 0; nt < 8; ++nt) {
            v16h vb = load_b_frag(Vb, T_, n0 + (nt << 4), s0, lane);
            acc[nt] = __builtin_amdgcn_wmma_f32_16x16x32_f16(false, a, false, vb,
                                                             (short)0, acc[nt], false, false);
        }
    }

    float* Ob = out + (long)b * T_ * DO_;
    const int mb = r0 + ((lane >> 4) << 3);
#pragma unroll
    for (int nt = 0; nt < 8; ++nt) {
        const int n = n0 + (nt << 4) + (lane & 15);
#pragma unroll
        for (int r = 0; r < 8; ++r)
            Ob[(long)(mb + r) * DO_ + n] = acc[nt][r];
    }
}

// ---------------------------------------------------------------------------
extern "C" void kernel_launch(void* const* d_in, const int* in_sizes, int n_in,
                              void* d_out, int out_size, void* d_ws, size_t ws_size,
                              hipStream_t stream) {
    (void)in_sizes; (void)n_in; (void)out_size; (void)ws_size;
    const float* x  = (const float*)d_in[0];
    const float* Wq = (const float*)d_in[1];
    const float* bq = (const float*)d_in[2];
    const float* Wk = (const float*)d_in[3];
    const float* bk = (const float*)d_in[4];
    const float* Wv = (const float*)d_in[5];
    const float* bv = (const float*)d_in[6];

    const int NX = B_ * T_ * D_;         // 16,777,216
    const int NW = D_ * D_;              //  1,048,576

    _Float16* ws  = (_Float16*)d_ws;
    _Float16* Xh  = ws;
    _Float16* WqT = Xh  + NX;
    _Float16* WkT = WqT + NW;
    _Float16* WvT = WkT + NW;
    _Float16* Qh  = WvT + NW;
    _Float16* Kh  = Qh  + NX;
    _Float16* Vt  = Kh  + NX;            // [B][DO][T]

    // 1) fp32 -> f16 conversions (weights transposed for B-fragment layout)
    cvt_f32_to_f16<<<NX / (256 * 8), 256, 0, stream>>>(x, Xh, NX);
    cvt_transpose_f32_to_f16<<<NW / 256, 256, 0, stream>>>(Wq, WqT, D_, D_);
    cvt_transpose_f32_to_f16<<<NW / 256, 256, 0, stream>>>(Wk, WkT, D_, D_);
    cvt_transpose_f32_to_f16<<<NW / 256, 256, 0, stream>>>(Wv, WvT, D_, DO_);

    // 2) projections: Q (scaled by 1/sqrt(D)), K, V (stored transposed)
    const int strips = (B_ * T_ / 16) * (D_ / 128);  // 8192
    const int blocks = strips / 4;                   // 2048
    proj_gemm_f16<<<blocks, 128, 0, stream>>>(Xh, WqT, bq, Qh, D_, D_, 0.03125f, 0);
    proj_gemm_f16<<<blocks, 128, 0, stream>>>(Xh, WkT, bk, Kh, D_, D_, 1.0f, 0);
    proj_gemm_f16<<<blocks, 128, 0, stream>>>(Xh, WvT, bv, Vt, D_, DO_, 1.0f, 1);

    // 3) attention: dynamic LDS = 64KB scores + 32KB Q tile
    const size_t shmem = (size_t)(16 * T_ + 16 * D_) * sizeof(_Float16);
    attn_kernel<<<dim3(T_ / 16, B_), 256, shmem, stream>>>(Qh, Kh, Vt, (float*)d_out);
}
